// Deform_GAU_14345190768898
// MI455X (gfx1250) — matrix operand order repaired and checked by reading, not compiled, not run on hardware
//
#include <hip/hip_runtime.h>

// ---------------- problem dims (fixed by reference) ----------------
#define DEV_N 4
#define C_HI  512
#define C_LO  256
#define H_HI  32
#define H_LO  64
#define HW    (H_LO * H_LO)     // 4096
#define K_LOW (9 * C_LO)        // 2304  (3x3 conv, K = tap*C + c)
#define K_UP  (4 * C_HI)        // 2048  (convT 4x4 s2 -> 4 taps per parity)

typedef _Float16 h8   __attribute__((ext_vector_type(8)));
typedef _Float16 v16h __attribute__((ext_vector_type(16)));
typedef float    v8f  __attribute__((ext_vector_type(8)));
typedef int      v4i  __attribute__((ext_vector_type(4)));

#define AS1 __attribute__((address_space(1)))
#define AS3 __attribute__((address_space(3)))

#if defined(__has_builtin)
#if __has_builtin(__builtin_amdgcn_global_load_async_to_lds_b128) && \
    __has_builtin(__builtin_amdgcn_s_wait_asynccnt)
#define USE_ASYNC_LDS 1
#endif
#endif
#ifndef USE_ASYNC_LDS
#define USE_ASYNC_LDS 0
#endif

// Stage one 16B chunk global->LDS (async when available); OOB -> zero-fill LDS.
__device__ __forceinline__ void stage16(const _Float16* gp, _Float16* lp, bool inb) {
  if (inb) {
#if USE_ASYNC_LDS
    // global src: flat address == global address -> int-to-AS1-pointer cast.
    // lds dst: generic ptr to __shared__ is {aperture_hi, lds_offset_lo};
    //          low 32 bits are exactly the AS3 offset.
    __builtin_amdgcn_global_load_async_to_lds_b128(
        (AS1 v4i*)(unsigned long long)(const void*)gp,
        (AS3 v4i*)(unsigned int)(unsigned long long)(const void*)lp,
        0, 0);
#else
    *(h8*)lp = *(const h8*)gp;
#endif
  } else {
    h8 z;
#pragma unroll
    for (int e = 0; e < 8; ++e) z[e] = (_Float16)0.f;
    *(h8*)lp = z;
  }
}

__device__ __forceinline__ void stage_fence() {
#if USE_ASYNC_LDS
  __builtin_amdgcn_s_wait_asynccnt(0);
#endif
  __syncthreads();
}

// =====================================================================
// Elementwise prep kernels
// =====================================================================

// NCHW f32 -> NHWC f16
__global__ void k_nchw_to_nhwc_h(const float* __restrict__ in, _Float16* __restrict__ out,
                                 int C, int H, int W) {
  const long total = (long)DEV_N * C * H * W;
  for (long i = (long)blockIdx.x * blockDim.x + threadIdx.x; i < total;
       i += (long)gridDim.x * blockDim.x) {
    int w = (int)(i % W);
    long t = i / W;
    int h = (int)(t % H); t /= H;
    int c = (int)(t % C);
    int n = (int)(t / C);
    out[(((long)n * H + h) * W + w) * C + c] = (_Float16)in[i];
  }
}

// w[OC][Cin][3][3] f32 -> wt[K=tap*Cin+c][OCpad] f16 (zero-padded cols)
__global__ void k_prep_w3x3(const float* __restrict__ w, _Float16* __restrict__ wt,
                            int Cin, int OCsrc, int OCpad) {
  const long total = (long)9 * Cin * OCpad;
  for (long i = (long)blockIdx.x * blockDim.x + threadIdx.x; i < total;
       i += (long)gridDim.x * blockDim.x) {
    int  oc = (int)(i % OCpad);
    long k  = i / OCpad;
    int  c  = (int)(k % Cin);
    int  kk = (int)(k / Cin);
    wt[i] = (oc < OCsrc) ? (_Float16)w[((long)oc * Cin + c) * 9 + kk] : (_Float16)0.f;
  }
}

// p_w[18][256][3][3] + m_w[9][256][3][3] -> wt[K_LOW][32] (cols 0..17 offset, 18..26 mask)
__global__ void k_prep_w_pm(const float* __restrict__ pw, const float* __restrict__ mw,
                            _Float16* __restrict__ wt) {
  const long total = (long)K_LOW * 32;
  for (long i = (long)blockIdx.x * blockDim.x + threadIdx.x; i < total;
       i += (long)gridDim.x * blockDim.x) {
    int  oc = (int)(i % 32);
    long k  = i / 32;
    int  c  = (int)(k % C_LO);
    int  kk = (int)(k / C_LO);
    float v = 0.f;
    if (oc < 18)      v = pw[((long)oc * C_LO + c) * 9 + kk];
    else if (oc < 27) v = mw[((long)(oc - 18) * C_LO + c) * 9 + kk];
    wt[i] = (_Float16)v;
  }
}

// up_w[I=512][O=256][4][4] -> wt[pyr][pxr][K=tap*512+ic][256]
// ConvTranspose2d(k=4,s=2,p=1): out[o] uses k with k%2 == (o+1)%2, i = (o+1-k)/2
__global__ void k_prep_w_up(const float* __restrict__ w, _Float16* __restrict__ wt) {
  const long total = (long)4 * K_UP * C_LO;
  for (long i = (long)blockIdx.x * blockDim.x + threadIdx.x; i < total;
       i += (long)gridDim.x * blockDim.x) {
    int  oc  = (int)(i % C_LO);
    long t2  = i / C_LO;
    int  k   = (int)(t2 % K_UP);
    int  par = (int)(t2 / K_UP);       // pyr*2 + pxr
    int  pxr = par & 1, pyr = par >> 1;
    int  ic  = k % C_HI;
    int  tap = k / C_HI;
    int  kyH = tap >> 1, kxH = tap & 1;
    int  ky  = (1 - pyr) + 2 * kyH;
    int  kx  = (1 - pxr) + 2 * kxH;
    wt[i] = (_Float16)w[(((long)ic * C_LO + oc) * 4 + ky) * 4 + kx];
  }
}

// =====================================================================
// WMMA implicit-GEMM 3x3 conv (pad=1), LDS-staged halo tile.
// act NHWC f16 [4,64,64,256], wt [K_LOW][OCpad] f16,
// out NCHW f32 [4,OCvalid,64,64].
// Block: WPB waves; all waves share one 16-pixel row tile (LDS halo),
// wave w computes oc slice [w*16*JC, (w+1)*16*JC).  WPB*JC*16 == OCpad.
// =====================================================================
template <int JC, int WPB>
__global__ void k_igemm_conv3x3(const _Float16* __restrict__ act,
                                const _Float16* __restrict__ wt,
                                float* __restrict__ out,
                                int OCpad, int OCvalid) {
  __shared__ _Float16 tile[3 * 18 * C_LO];   // 27,648 B halo: 3 rows x 18 cols x 256ch

  const int tid  = threadIdx.x;
  const int lane = tid & 31;
  const int wv   = tid >> 5;
  const int M    = lane & 15;
  const int half = lane >> 4;
  const int rt   = blockIdx.x;               // 1024 row tiles
  const int xb   = rt & 3;
  const int y    = (rt >> 2) & 63;
  const int n    = rt >> 8;
  const int oc0  = wv * (16 * JC);

  // ---- cooperative halo staging (async global->LDS, zero-fill OOB) ----
  for (int u = tid; u < 3 * 18 * 32; u += 32 * WPB) {
    const int cu = u & 31;                   // 8-channel block
    const int xc = (u >> 5) % 18;            // halo x: 0..17
    const int r  = (u >> 5) / 18;            // halo y: 0..2
    const int yy = y + r - 1;
    const int xx = xb * 16 + xc - 1;
    const bool inb = (yy >= 0) & (yy < H_LO) & (xx >= 0) & (xx < H_LO);
    const _Float16* gp =
        act + ((((long)(n * H_LO + (inb ? yy : 0)) * H_LO + (inb ? xx : 0)) << 8) + (cu << 3));
    stage16(gp, &tile[((r * 18 + xc) << 8) + (cu << 3)], inb);
  }
  stage_fence();

  v8f acc[JC];
#pragma unroll
  for (int j = 0; j < JC; ++j)
#pragma unroll
    for (int e = 0; e < 8; ++e) acc[j][e] = 0.f;

  for (int kk = 0; kk < 9; ++kk) {
    const _Float16* tp = &tile[((kk / 3) * 18 + M + (kk % 3)) << 8];
#pragma unroll 2
    for (int cb = 0; cb < C_LO; cb += 32) {
      h8 r0 = *(const h8*)(tp + cb + half * 8);        // ds_load_b128
      h8 r1 = *(const h8*)(tp + cb + 16 + half * 8);   // ds_load_b128
      v16h a;
#pragma unroll
      for (int e = 0; e < 8; ++e) { a[e] = r0[e]; a[8 + e] = r1[e]; }
      const int k0 = kk * C_LO + cb;
      const _Float16* bp = wt + (long)k0 * OCpad + oc0 + (lane & 15);
#pragma unroll
      for (int j = 0; j < JC; ++j) {
        v16h b;
#pragma unroll
        for (int s = 0; s < 16; ++s) b[s] = bp[(long)((s & 7) + half * 8 + (s >> 3) * 16) * OCpad + j * 16];
        acc[j] = __builtin_amdgcn_wmma_f32_16x16x32_f16(false, a, false, b,
                                                        (short)0, acc[j], false, false);
      }
    }
  }

  const int xs0 = xb * 16 + 8 * half;        // D elem r -> pixel M = r + 8*half
#pragma unroll
  for (int j = 0; j < JC; ++j) {
    const int oc = oc0 + j * 16 + (lane & 15);
    if (oc < OCvalid) {
      float* op = out + (((long)(n * OCvalid + oc) * H_LO + y) * H_LO + xs0);
#pragma unroll
      for (int r = 0; r < 8; ++r) op[r] = acc[j][r];
    }
  }
}

// =====================================================================
// WMMA conv-transpose 4x4 s2 p1 : act NHWC f16 [4,32,32,512],
// wt [2][2][K_UP][256] f16, out NCHW f32 [4,256,64,64].
// Pixel tiles are 16 same-parity x positions in one output row so the
// weight parity class is uniform across the wave.
// =====================================================================
__global__ void k_igemm_convtr(const _Float16* __restrict__ act,
                               const _Float16* __restrict__ wt,
                               float* __restrict__ out) {
  const int lane  = threadIdx.x;
  const int M     = lane & 15;
  const int half  = lane >> 4;
  const int rt    = blockIdx.x;           // 1024 tiles
  const int xblk  = rt & 1;
  const int xpar  = (rt >> 1) & 1;
  const int oy    = (rt >> 2) & 63;
  const int n     = rt >> 8;
  const int q     = xblk * 16 + M;        // 0..31 (x index within parity class)
  const int oyPar = oy & 1;
  const int oc0   = blockIdx.y * 64;
  const _Float16* wbase = wt + (long)(oyPar * 2 + xpar) * K_UP * C_LO;

  v8f acc[4];
#pragma unroll
  for (int j = 0; j < 4; ++j)
#pragma unroll
    for (int e = 0; e < 8; ++e) acc[j][e] = 0.f;

  for (int tap = 0; tap < 4; ++tap) {
    const int kyH = tap >> 1, kxH = tap & 1;
    const int iy  = ((oy + oyPar) >> 1) - kyH;
    const int ix  = xpar + q - kxH;
    const bool inb = (iy >= 0) & (iy < H_HI) & (ix >= 0) & (ix < H_HI);
    const _Float16* ap =
        act + (((long)(n * H_HI + (inb ? iy : 0)) * H_HI + (inb ? ix : 0)) * C_HI);
#pragma unroll 2
    for (int cb = 0; cb < C_HI; cb += 32) {
      v16h a;
      if (inb) {
        h8 r0 = *(const h8*)(ap + cb + half * 8);
        h8 r1 = *(const h8*)(ap + cb + 16 + half * 8);
#pragma unroll
        for (int e = 0; e < 8; ++e) { a[e] = r0[e]; a[8 + e] = r1[e]; }
      } else {
#pragma unroll
        for (int e = 0; e < 16; ++e) a[e] = (_Float16)0.f;
      }
      const int k0 = tap * C_HI + cb;
      const _Float16* bp = wbase + (long)k0 * C_LO + oc0 + (lane & 15);
#pragma unroll
      for (int j = 0; j < 4; ++j) {
        v16h b;
#pragma unroll
        for (int s = 0; s < 16; ++s) b[s] = bp[(long)((s & 7) + half * 8 + (s >> 3) * 16) * C_LO + j * 16];
        acc[j] = __builtin_amdgcn_wmma_f32_16x16x32_f16(false, a, false, b,
                                                        (short)0, acc[j], false, false);
      }
    }
  }

#pragma unroll
  for (int j = 0; j < 4; ++j) {
    const int oc = oc0 + j * 16 + (lane & 15);
    float* op = out + (((long)(n * C_LO + oc) * H_LO + oy) * H_LO);
#pragma unroll
    for (int r = 0; r < 8; ++r) {
      const int xo = xpar + 2 * (xblk * 16 + 8 * half + r);
      op[xo] = acc[j][r];
    }
  }
}

// =====================================================================
// Batchnorm statistics (training mode, biased var) over (N,H,W) per channel
// =====================================================================
__global__ void k_bnstats(const float* __restrict__ data, float* __restrict__ mean,
                          float* __restrict__ rstd) {
  const int c = blockIdx.x;
  const int tid = threadIdx.x;
  float s = 0.f, sq = 0.f;
  for (int i = tid; i < DEV_N * HW; i += 256) {
    const int n = i >> 12, p = i & (HW - 1);
    const float v = data[(long)(n * C_LO + c) * HW + p];
    s += v; sq += v * v;
  }
  __shared__ float ssum[256], ssq[256];
  ssum[tid] = s; ssq[tid] = sq;
  __syncthreads();
  for (int st = 128; st > 0; st >>= 1) {
    if (tid < st) { ssum[tid] += ssum[tid + st]; ssq[tid] += ssq[tid + st]; }
    __syncthreads();
  }
  if (tid == 0) {
    const float inv = 1.f / (float)(DEV_N * HW);
    const float mu  = ssum[0] * inv;
    const float var = ssq[0] * inv - mu * mu;
    mean[c] = mu;
    rstd[c] = rsqrtf(var + 1e-5f);
  }
}

// Apply BN to NCHW f32, emit NHWC f16
__global__ void k_norm_to_nhwc(const float* __restrict__ raw, const float* __restrict__ mean,
                               const float* __restrict__ rstd, const float* __restrict__ g,
                               const float* __restrict__ b, _Float16* __restrict__ outh) {
  const long total = (long)DEV_N * C_LO * HW;
  for (long i = (long)blockIdx.x * blockDim.x + threadIdx.x; i < total;
       i += (long)gridDim.x * blockDim.x) {
    const int p = (int)(i & (HW - 1));
    const int c = (int)((i >> 12) & (C_LO - 1));
    const int n = (int)(i >> 20);
    const float v = (raw[i] - mean[c]) * rstd[c] * g[c] + b[c];
    const int y = p >> 6, x = p & 63;
    outh[(((long)(n * H_LO + y)) * H_LO + x) * C_LO + c] = (_Float16)v;
  }
}

// =====================================================================
// Modulated deformable 3x3 conv as WMMA GEMM with fused bilinear A-builder
// =====================================================================
__device__ __forceinline__ h8 ld8(const _Float16* p, bool v) {
  h8 r;
  if (v) {
    r = *(const h8*)p;
  } else {
#pragma unroll
    for (int e = 0; e < 8; ++e) r[e] = (_Float16)0.f;
  }
  return r;
}

__global__ void k_igemm_dconv(const _Float16* __restrict__ act,   // highn NHWC f16
                              const _Float16* __restrict__ wt,    // [K_LOW][256] f16
                              const float* __restrict__ pm,       // NCHW [4,27,64,64]
                              const float* __restrict__ pb,       // [18]
                              const float* __restrict__ mb,       // [9]
                              float* __restrict__ out) {          // NCHW [4,256,64,64]
  const int lane = threadIdx.x;
  const int M    = lane & 15;
  const int half = lane >> 4;
  const int rt   = blockIdx.x;
  const int xb   = rt & 3;
  const int y    = (rt >> 2) & 63;
  const int n    = rt >> 8;
  const int x    = xb * 16 + M;
  const int p    = y * H_LO + x;
  const int oc0  = blockIdx.y * 64;

  v8f acc[4];
#pragma unroll
  for (int j = 0; j < 4; ++j)
#pragma unroll
    for (int e = 0; e < 8; ++e) acc[j][e] = 0.f;

  for (int kk = 0; kk < 9; ++kk) {
    const int ky = kk / 3 - 1, kx = kk % 3 - 1;
    const float dy = pm[(long)(n * 27 + kk) * HW + p] + pb[kk];
    const float dx = pm[(long)(n * 27 + 9 + kk) * HW + p] + pb[9 + kk];
    const float mr = pm[(long)(n * 27 + 18 + kk) * HW + p] + mb[kk];
    const float mg = 1.f / (1.f + __expf(-mr));     // modulation gate
    float py = (float)(y + 1 + ky) + dy;            // padded coords, pad=1
    float px = (float)(x + 1 + kx) + dx;
    py = fminf(fmaxf(py, 0.f), 65.f);
    px = fminf(fmaxf(px, 0.f), 65.f);
    const float y0f = floorf(py), x0f = floorf(px);
    const int y0 = (int)y0f, x0 = (int)x0f;
    const int y1 = min(y0 + 1, 65), x1 = min(x0 + 1, 65);
    const float wy = py - y0f, wx = px - x0f;
    const float w00 = (1.f - wy) * (1.f - wx);
    const float w01 = (1.f - wy) * wx;
    const float w10 = wy * (1.f - wx);
    const float w11 = wy * wx;
    // padded coord q -> source q-1, valid in [0,64); OOB samples are zero
    const int sy0 = y0 - 1, sy1 = y1 - 1, sx0 = x0 - 1, sx1 = x1 - 1;
    const bool vy0 = (sy0 >= 0) & (sy0 < H_LO);
    const bool vy1 = (sy1 >= 0) & (sy1 < H_LO);
    const bool vx0 = (sx0 >= 0) & (sx0 < H_LO);
    const bool vx1 = (sx1 >= 0) & (sx1 < H_LO);
    const long base = (long)n * H_LO * H_LO * C_LO;
    const _Float16* p00 = act + base + (long)((vy0 ? sy0 : 0) * H_LO + (vx0 ? sx0 : 0)) * C_LO;
    const _Float16* p01 = act + base + (long)((vy0 ? sy0 : 0) * H_LO + (vx1 ? sx1 : 0)) * C_LO;
    const _Float16* p10 = act + base + (long)((vy1 ? sy1 : 0) * H_LO + (vx0 ? sx0 : 0)) * C_LO;
    const _Float16* p11 = act + base + (long)((vy1 ? sy1 : 0) * H_LO + (vx1 ? sx1 : 0)) * C_LO;
    const bool b00 = vy0 & vx0, b01 = vy0 & vx1, b10 = vy1 & vx0, b11 = vy1 & vx1;

    for (int cb = 0; cb < C_LO; cb += 32) {
      v16h a;
#pragma unroll
      for (int run = 0; run < 2; ++run) {
        const int co = cb + run * 16 + half * 8;
        h8 s00 = ld8(p00 + co, b00);
        h8 s01 = ld8(p01 + co, b01);
        h8 s10 = ld8(p10 + co, b10);
        h8 s11 = ld8(p11 + co, b11);
#pragma unroll
        for (int e = 0; e < 8; ++e) {
          const float v = w00 * (float)s00[e] + w01 * (float)s01[e] +
                          w10 * (float)s10[e] + w11 * (float)s11[e];
          a[run * 8 + e] = (_Float16)(v * mg);
        }
      }
      const int k0 = kk * C_LO + cb;
      const _Float16* bp = wt + (long)k0 * C_LO + oc0 + (lane & 15);
#pragma unroll
      for (int j = 0; j < 4; ++j) {
        v16h b;
#pragma unroll
        for (int s = 0; s < 16; ++s) b[s] = bp[(long)((s & 7) + half * 8 + (s >> 3) * 16) * C_LO + j * 16];
        acc[j] = __builtin_amdgcn_wmma_f32_16x16x32_f16(false, a, false, b,
                                                        (short)0, acc[j], false, false);
      }
    }
  }

  const int xs0 = xb * 16 + 8 * half;
#pragma unroll
  for (int j = 0; j < 4; ++j) {
    const int oc = oc0 + j * 16 + (lane & 15);
    float* op = out + (((long)(n * C_LO + oc) * H_LO + y) * H_LO + xs0);
#pragma unroll
    for (int r = 0; r < 8; ++r) op[r] = acc[j][r];
  }
}

// sigmoid(mean over 9 mask channels + bias) per pixel
__global__ void k_mask_mean(const float* __restrict__ pm, const float* __restrict__ mb,
                            float* __restrict__ mask) {
  const int total = DEV_N * HW;
  for (int i = blockIdx.x * blockDim.x + threadIdx.x; i < total;
       i += gridDim.x * blockDim.x) {
    const int p = i & (HW - 1), n = i >> 12;
    float s = 0.f;
#pragma unroll
    for (int k = 0; k < 9; ++k) s += pm[(long)(n * 27 + 18 + k) * HW + p] + mb[k];
    const float mr = s * (1.f / 9.f);
    mask[i] = 1.f / (1.f + __expf(-mr));
  }
}

// out = relu( BN(low) * mask + BN(d) )
__global__ void k_final(const float* __restrict__ lowraw, const float* __restrict__ draw,
                        const float* __restrict__ mean_l, const float* __restrict__ rstd_l,
                        const float* __restrict__ gl, const float* __restrict__ bl,
                        const float* __restrict__ mean_d, const float* __restrict__ rstd_d,
                        const float* __restrict__ gh, const float* __restrict__ bh,
                        const float* __restrict__ mask, float* __restrict__ out) {
  const long total = (long)DEV_N * C_LO * HW;
  for (long i = (long)blockIdx.x * blockDim.x + threadIdx.x; i < total;
       i += (long)gridDim.x * blockDim.x) {
    const int p = (int)(i & (HW - 1));
    const int c = (int)((i >> 12) & (C_LO - 1));
    const int n = (int)(i >> 20);
    const float lv = (lowraw[i] - mean_l[c]) * rstd_l[c] * gl[c] + bl[c];
    const float dv = (draw[i]  - mean_d[c]) * rstd_d[c] * gh[c] + bh[c];
    const float r  = lv * mask[n * HW + p] + dv;
    out[i] = fmaxf(r, 0.f);
  }
}

// =====================================================================
// Launcher
// =====================================================================
extern "C" void kernel_launch(void* const* d_in, const int* in_sizes, int n_in,
                              void* d_out, int out_size, void* d_ws, size_t ws_size,
                              hipStream_t stream) {
  const float* fms_high  = (const float*)d_in[0];
  const float* fms_low   = (const float*)d_in[1];
  const float* conv3x3_w = (const float*)d_in[2];
  const float* bn_low_g  = (const float*)d_in[3];
  const float* bn_low_b  = (const float*)d_in[4];
  const float* up_w      = (const float*)d_in[5];
  const float* bn_up_g   = (const float*)d_in[6];
  const float* bn_up_b   = (const float*)d_in[7];
  const float* p_w       = (const float*)d_in[8];
  const float* p_b       = (const float*)d_in[9];
  const float* m_w       = (const float*)d_in[10];
  const float* m_b       = (const float*)d_in[11];
  const float* dconv_w   = (const float*)d_in[12];
  const float* bn_high_g = (const float*)d_in[13];
  const float* bn_high_b = (const float*)d_in[14];
  float* out = (float*)d_out;

  char* ws = (char*)d_ws;
  size_t off = 0;
  auto alloc = [&](size_t bytes) -> char* {
    char* pp = ws + off;
    off = (off + bytes + 255) & ~(size_t)255;
    return pp;
  };
  _Float16* lowh  = (_Float16*)alloc((size_t)DEV_N * HW * C_LO * 2);          // fms_low NHWC f16
  _Float16* highh = (_Float16*)alloc((size_t)DEV_N * H_HI * H_HI * C_HI * 2); // fms_high NHWC f16
  _Float16* highn = (_Float16*)alloc((size_t)DEV_N * HW * C_LO * 2);          // BN'd upsample NHWC f16
  _Float16* Wlow  = (_Float16*)alloc((size_t)K_LOW * C_LO * 2);
  _Float16* Wd    = (_Float16*)alloc((size_t)K_LOW * C_LO * 2);
  _Float16* Wpm   = (_Float16*)alloc((size_t)K_LOW * 32 * 2);
  _Float16* Wup   = (_Float16*)alloc((size_t)4 * K_UP * C_LO * 2);
  float* upraw  = (float*)alloc((size_t)DEV_N * C_LO * HW * 4);
  float* lowraw = (float*)alloc((size_t)DEV_N * C_LO * HW * 4);
  float* draw   = (float*)alloc((size_t)DEV_N * C_LO * HW * 4);
  float* pmraw  = (float*)alloc((size_t)DEV_N * 27 * HW * 4);
  float* maskb  = (float*)alloc((size_t)DEV_N * HW * 4);
  float* stats  = (float*)alloc((size_t)6 * C_LO * 4);
  float* mean_up = stats,            * rstd_up = stats + C_LO;
  float* mean_l  = stats + 2 * C_LO, * rstd_l  = stats + 3 * C_LO;
  float* mean_d  = stats + 4 * C_LO, * rstd_d  = stats + 5 * C_LO;

  const dim3 b256(256);
  const dim3 wave(32);

  // --- prep: layout conversions + weight transposes (f32 -> f16) ---
  k_nchw_to_nhwc_h<<<2048, b256, 0, stream>>>(fms_low, lowh, C_LO, H_LO, H_LO);
  k_nchw_to_nhwc_h<<<2048, b256, 0, stream>>>(fms_high, highh, C_HI, H_HI, H_HI);
  k_prep_w3x3<<<1024, b256, 0, stream>>>(conv3x3_w, Wlow, C_LO, C_LO, C_LO);
  k_prep_w3x3<<<1024, b256, 0, stream>>>(dconv_w, Wd, C_LO, C_LO, C_LO);
  k_prep_w_pm<<<256, b256, 0, stream>>>(p_w, m_w, Wpm);
  k_prep_w_up<<<2048, b256, 0, stream>>>(up_w, Wup);

  // --- high path: conv-transpose (WMMA), BN stats, normalize -> NHWC f16 ---
  k_igemm_convtr<<<dim3(1024, 4), wave, 0, stream>>>(highh, Wup, upraw);
  k_bnstats<<<C_LO, b256, 0, stream>>>(upraw, mean_up, rstd_up);
  k_norm_to_nhwc<<<2048, b256, 0, stream>>>(upraw, mean_up, rstd_up, bn_up_g, bn_up_b, highn);

  // --- offset + mask convs fused as one 27-col WMMA GEMM (1 wave/block) ---
  k_igemm_conv3x3<2, 1><<<dim3(1024), dim3(32), 0, stream>>>(highn, Wpm, pmraw, 32, 27);

  // --- low path conv (WMMA, 4 waves sharing one LDS halo tile) ---
  k_igemm_conv3x3<4, 4><<<dim3(1024), dim3(128), 0, stream>>>(lowh, Wlow, lowraw, C_LO, C_LO);

  // --- modulated deformable conv (bilinear fused into WMMA A-fragments) ---
  k_igemm_dconv<<<dim3(1024, 4), wave, 0, stream>>>(highn, Wd, pmraw, p_b, m_b, draw);

  // --- stats + fused epilogue ---
  k_bnstats<<<C_LO, b256, 0, stream>>>(lowraw, mean_l, rstd_l);
  k_bnstats<<<C_LO, b256, 0, stream>>>(draw, mean_d, rstd_d);
  k_mask_mean<<<64, b256, 0, stream>>>(pmraw, m_b, maskb);
  k_final<<<4096, b256, 0, stream>>>(lowraw, draw, mean_l, rstd_l, bn_low_g, bn_low_b,
                                     mean_d, rstd_d, bn_high_g, bn_high_b, maskb, out);

  (void)in_sizes; (void)n_in; (void)out_size; (void)ws_size;
}
